// CustomLSTM_29506425323624
// MI455X (gfx1250) — matrix operand order repaired and checked by reading, not compiled, not run on hardware
//
#include <hip/hip_runtime.h>

// ---------------------------------------------------------------------------
// LSTM  (B=128, S=1024, I=128, H=128) for gfx1250 (MI455X, wave32, WMMA)
//
// Phase 1: xp[s,b,g] = x[b,s,:] . W_ih[g,:] + b_ih[g] + b_hh[g]  (bf16 WMMA,
//          f32 accum; TIME-MAJOR layout so the scan reads contiguous slabs)
// Phase 2: sequential scan; 8 workgroups (16 batch rows each), W_hh resident
//          in VGPRs as bf16 B-fragments, h broadcast via LDS, c in registers,
//          xp slabs double-buffered in LDS via async global->LDS DMA
//          (ASYNCcnt) issued one timestep ahead.  xp is added in the
//          elementwise phase so WMMAs depend only on the h fragments.
// ---------------------------------------------------------------------------

typedef __attribute__((ext_vector_type(16))) __bf16 v16bf;
typedef __attribute__((ext_vector_type(8)))  float  v8f;

#define BDIM 128
#define SDIM 1024
#define IDIM 128
#define HDIM 128
#define GDIM 512           // 4*H
#define GP   (GDIM + 4)    // padded gates row (bank-conflict avoidance)
#define XPP  (GDIM + 4)    // padded staged-xp row
#define HP   (HDIM + 8)    // padded h row

__device__ __forceinline__ float fast_sigmoid(float x) {
    return __builtin_amdgcn_rcpf(1.0f + __expf(-x));
}
__device__ __forceinline__ float fast_tanh(float x) {
#if __has_builtin(__builtin_amdgcn_tanhf)
    return __builtin_amdgcn_tanhf(x);
#else
    x = fminf(fmaxf(x, -15.0f), 15.0f);
    float e = __expf(2.0f * x);
    return (e - 1.0f) * __builtin_amdgcn_rcpf(e + 1.0f);
#endif
}

// ---------------------------------------------------------------------------
// Phase 1: input projection GEMM.  Block = 256 threads = 8 waves.
// Block covers M=16 rows of (B*S), full N=512 (wave w -> cols [w*64, w*64+64)).
// Output stored time-major: xp[(s*B + b)*G + g].
// ---------------------------------------------------------------------------
__global__ __launch_bounds__(256)
void lstm_xproj_kernel(const float* __restrict__ x,     // [B*S, I] (b-major)
                       const float* __restrict__ W_ih,  // [G, I]
                       const float* __restrict__ b_ih,  // [G]
                       const float* __restrict__ b_hh,  // [G]
                       float* __restrict__ xp)          // [S, B, G] time-major
{
    const int lane = threadIdx.x & 31;
    const int wave = threadIdx.x >> 5;          // 0..7
    const int m0   = blockIdx.x * 16;           // row tile in B*S (b-major)
    const int n0   = wave * 64;                 // 4 N-tiles of 16

    const int mrow = lane & 15;
    const int half = lane >> 4;                 // 0: lanes 0-15, 1: lanes 16-31
    const int ncol = lane & 15;
    const int kh   = half * 16;

    // A fragments (x rows, f32 -> bf16). Contiguous K layout (matches B):
    //   lanes 0-15 : K = k0+0..15
    //   lanes 16-31: K = k0+16..31
    v16bf a[4];
    const float* xrow = x + (size_t)(m0 + mrow) * IDIM;
#pragma unroll
    for (int kf = 0; kf < 4; ++kf) {
        const float* p1 = xrow + kf * 32 + half * 16;
        const float* p2 = xrow + kf * 32 + half * 16 + 8;
#pragma unroll
        for (int e = 0; e < 8; ++e) {
            a[kf][e]     = (__bf16)p1[e];
            a[kf][8 + e] = (__bf16)p2[e];
        }
    }

#pragma unroll
    for (int t = 0; t < 4; ++t) {
        const int n = n0 + t * 16 + ncol;       // this lane's output column
        const float bias = b_ih[n] + b_hh[n];
        v8f acc;
#pragma unroll
        for (int v = 0; v < 8; ++v) acc[v] = bias;

        // B fragments: B[k][n] = W_ih[n][k]; lane holds column n,
        // K = k0+kh+0..15 packed 2/VGPR.
        const float* wrow = W_ih + (size_t)n * IDIM;
#pragma unroll
        for (int kf = 0; kf < 4; ++kf) {
            v16bf b;
            const float* p = wrow + kf * 32 + kh;
#pragma unroll
            for (int e = 0; e < 16; ++e) b[e] = (__bf16)p[e];
            acc = __builtin_amdgcn_wmma_f32_16x16x32_bf16(
                false, a[kf], false, b, (short)0, acc, false, false);
        }

        // C/D layout: VGPR v -> row v + 8*half, lane%16 -> column.
        // Scatter into time-major xp: row mg = b*S+s  ->  (s*B + b).
#pragma unroll
        for (int v = 0; v < 8; ++v) {
            const int mg = m0 + v + 8 * half;
            const int bb = mg >> 10;            // /S
            const int ss = mg & (SDIM - 1);     // %S
            xp[((size_t)ss * BDIM + bb) * GDIM + n] = acc[v];
        }
    }
}

// ---------------------------------------------------------------------------
// Phase 2: recurrence.  Grid = 8 blocks (16 batch rows each), 512 thr = 16 waves.
// Wave w owns gate columns [w*32, w*32+32) (2 N-tiles), W_hh fragments in VGPRs.
// xp slabs (32 KB/step/block, contiguous in time-major layout) are DMA'd into
// LDS with global_load_async_to_lds_b128, double buffered, one step ahead.
// ---------------------------------------------------------------------------
#define SCAN_THREADS 512

// Issue async copy of one step's xp slab (16 rows x 512 f32, contiguous 32 KB
// in global) into a padded LDS buffer. 2048 16-byte chunks over 512 threads.
__device__ __forceinline__ void issue_xp_async(const float* __restrict__ xp,
                                               int s, int b0,
                                               float* dstbase, int tid) {
    const float* g0 = xp + ((size_t)s * BDIM + b0) * GDIM;
#pragma unroll
    for (int k = 0; k < 4; ++k) {
        const int chunk = tid + k * SCAN_THREADS;        // 0..2047
        const int row   = chunk >> 7;                    // 2048B (=G floats)/row
        const int cw    = chunk & 127;                   // 16B chunk in row
        const unsigned lds =
            (unsigned)(size_t)(dstbase + (size_t)row * XPP) + (unsigned)cw * 16u;
        const unsigned long long ga =
            (unsigned long long)(size_t)(g0 + (size_t)chunk * 4);
        asm volatile("global_load_async_to_lds_b128 %0, %1, off"
                     :: "v"(lds), "v"(ga) : "memory");
    }
}

__device__ __forceinline__ void wait_async_zero() {
    asm volatile("s_wait_asynccnt 0x0" ::: "memory");
}

__global__ __launch_bounds__(SCAN_THREADS)
void lstm_scan_kernel(const float* __restrict__ xp,     // [S, B, G]
                      const float* __restrict__ W_hh,   // [G, H]
                      float* __restrict__ out)          // outputs|h_T|c_T
{
    __shared__ float  xps[2][16][XPP];      // double-buffered xp_t slab
    __shared__ float  gates_lds[16][GP];    // h @ W_hh^T partial gates
    __shared__ __bf16 h_lds[16][HP];        // current h (bf16, row-major)

    const int tid  = threadIdx.x;
    const int lane = tid & 31;
    const int wave = tid >> 5;              // 0..15
    const int b0   = blockIdx.x * 16;       // batch row base

    const int mrow = lane & 15;
    const int half = lane >> 4;
    const int ncol = lane & 15;
    const int kh   = half * 16;

    // Prefetch step 0's xp slab while we set up weights.
    issue_xp_async(xp, 0, b0, &xps[0][0][0], tid);

    // Resident W_hh^T B-fragments: B[k][n] = W_hh[n][k], n in wave's 32 cols.
    v16bf bfrag[2][4];
#pragma unroll
    for (int t = 0; t < 2; ++t) {
        const int n = wave * 32 + t * 16 + ncol;
        const float* wrow = W_hh + (size_t)n * HDIM;
#pragma unroll
        for (int kf = 0; kf < 4; ++kf) {
#pragma unroll
            for (int e = 0; e < 16; ++e)
                bfrag[t][kf][e] = (__bf16)wrow[kf * 32 + kh + e];
        }
    }

    // Elementwise ownership: thread -> row er, 4 consecutive cols at ej.
    const int er = tid >> 5;                // 0..15
    const int ej = (tid & 31) * 4;          // 0..124

    float c_reg[4] = {0.f, 0.f, 0.f, 0.f};
    float hv[4]    = {0.f, 0.f, 0.f, 0.f};

    // h0 = 0
    {
        union { uint2 u; __bf16 b[4]; } hz;
#pragma unroll
        for (int q = 0; q < 4; ++q) hz.b[q] = (__bf16)0.0f;
        *(uint2*)&h_lds[er][ej] = hz.u;
    }
    wait_async_zero();          // step-0 slab landed (this wave's chunks)
    __syncthreads();            // ... and everyone else's

    for (int s = 0; s < SDIM; ++s) {
        const int cur = s & 1;

        // Kick off next step's slab DMA; overlaps with all compute below.
        if (s + 1 < SDIM)
            issue_xp_async(xp, s + 1, b0, &xps[cur ^ 1][0][0], tid);

        // --- A fragments of h from LDS (all waves read the same h tile) ---
        v16bf a[4];
#pragma unroll
        for (int kf = 0; kf < 4; ++kf) {
            const uint4* p1 = (const uint4*)&h_lds[mrow][kf * 32 + half * 16];
            const uint4* p2 = (const uint4*)&h_lds[mrow][kf * 32 + half * 16 + 8];
            union { v16bf v; uint4 q[2]; } u;
            u.q[0] = *p1;
            u.q[1] = *p2;
            a[kf] = u.v;
        }

        // --- partial gates = h @ W_hh^T (WMMA, zero-seeded accumulators) ---
#pragma unroll
        for (int t = 0; t < 2; ++t) {
            const int n = wave * 32 + t * 16 + ncol;
            v8f acc = {};
#pragma unroll
            for (int kf = 0; kf < 4; ++kf)
                acc = __builtin_amdgcn_wmma_f32_16x16x32_bf16(
                    false, a[kf], false, bfrag[t][kf], (short)0, acc, false, false);
#pragma unroll
            for (int v = 0; v < 8; ++v)
                gates_lds[v + 8 * half][n] = acc[v];
        }
        __syncthreads();   // gates visible; everyone done reading old h

        // --- elementwise: gates += xp_t, then i,f,g,o update (ref order) ---
        {
            const float4 pi = *(const float4*)&gates_lds[er][ej];
            const float4 pf = *(const float4*)&gates_lds[er][128 + ej];
            const float4 pg = *(const float4*)&gates_lds[er][256 + ej];
            const float4 po = *(const float4*)&gates_lds[er][384 + ej];
            const float4 xi = *(const float4*)&xps[cur][er][ej];
            const float4 xf = *(const float4*)&xps[cur][er][128 + ej];
            const float4 xg = *(const float4*)&xps[cur][er][256 + ej];
            const float4 xo = *(const float4*)&xps[cur][er][384 + ej];
            const float gi[4] = {pi.x + xi.x, pi.y + xi.y, pi.z + xi.z, pi.w + xi.w};
            const float gf[4] = {pf.x + xf.x, pf.y + xf.y, pf.z + xf.z, pf.w + xf.w};
            const float gg[4] = {pg.x + xg.x, pg.y + xg.y, pg.z + xg.z, pg.w + xg.w};
            const float go[4] = {po.x + xo.x, po.y + xo.y, po.z + xo.z, po.w + xo.w};
            union { uint2 u; __bf16 b[4]; } hb;
#pragma unroll
            for (int q = 0; q < 4; ++q) {
                const float ig = fast_sigmoid(gi[q]);
                const float fg = fast_sigmoid(gf[q]);
                const float g2 = fast_tanh   (gg[q]);
                const float og = fast_sigmoid(go[q]);
                const float cn = fg * c_reg[q] + ig * g2;
                c_reg[q] = cn;
                hv[q]    = og * fast_tanh(cn);
                hb.b[q]  = (__bf16)hv[q];
            }
            *(uint2*)&h_lds[er][ej] = hb.u;   // ds_store_b64
        }
        // outputs[b, s, :] (f32, coalesced 16B per thread)
        float4 o4 = make_float4(hv[0], hv[1], hv[2], hv[3]);
        *(float4*)(out + (((size_t)(b0 + er)) * SDIM + s) * HDIM + ej) = o4;

        if (s + 1 < SDIM) wait_async_zero();  // my next-slab chunks are in LDS
        __syncthreads();   // new h + next slab visible to all waves
    }

    // Tail: h_T then c_T, appended after outputs.
    const size_t oH = (size_t)BDIM * SDIM * HDIM;
    float4 h4 = make_float4(hv[0], hv[1], hv[2], hv[3]);
    float4 c4 = make_float4(c_reg[0], c_reg[1], c_reg[2], c_reg[3]);
    *(float4*)(out + oH + (size_t)(b0 + er) * HDIM + ej) = h4;
    *(float4*)(out + oH + (size_t)BDIM * HDIM + (size_t)(b0 + er) * HDIM + ej) = c4;
}

// ---------------------------------------------------------------------------
extern "C" void kernel_launch(void* const* d_in, const int* in_sizes, int n_in,
                              void* d_out, int out_size, void* d_ws, size_t ws_size,
                              hipStream_t stream) {
    const float* x    = (const float*)d_in[0];   // [B,S,I]
    const float* W_ih = (const float*)d_in[1];   // [4H,I]
    const float* W_hh = (const float*)d_in[2];   // [4H,H]
    const float* b_ih = (const float*)d_in[3];   // [4H]
    const float* b_hh = (const float*)d_in[4];   // [4H]
    float* out = (float*)d_out;
    float* xp  = (float*)d_ws;                   // [S, B, 4H] f32 = 256 MB scratch

    // Phase 1: 8192 blocks x 256 threads, each block does a 16x512 slab of xp.
    lstm_xproj_kernel<<<dim3((BDIM * SDIM) / 16), 256, 0, stream>>>(
        x, W_ih, b_ih, b_hh, xp);

    // Phase 2: 8 blocks (one per 16 batch rows) x 512 threads, sequential scan.
    lstm_scan_kernel<<<dim3(BDIM / 16), SCAN_THREADS, 0, stream>>>(
        xp, W_hh, out);
}
